// GatedSparseAttention_25640954757704
// MI455X (gfx1250) — matrix operand order repaired and checked by reading, not compiled, not run on hardware
//
#include <hip/hip_runtime.h>
#include <hip/hip_bf16.h>

typedef __attribute__((ext_vector_type(16))) __bf16 v16bf;
typedef __attribute__((ext_vector_type(8)))  __bf16 v8bf;
typedef __attribute__((ext_vector_type(2)))  __bf16 v2bf;
typedef __attribute__((ext_vector_type(8)))  float  v8f;

#define T_SEQ 1024
#define C_DIM 512
#define H_N   8
#define HD_N  64
#define HI_N  4
#define DI_N  32
#define KMAX  128
#define KMIN  32
#define KBASE 64
#define SINKN 4
#define NEGV  (-1e30f)
#define NPROJ 2560   // q|k|v|gv|go concatenated along N

__device__ __forceinline__ float bf2f(__bf16 v) { return (float)v; }

// ---------------- f32 -> bf16 convert ----------------
__global__ void k_f32_to_bf16(const float* __restrict__ s, __bf16* __restrict__ d, int n) {
  int i = blockIdx.x * blockDim.x + threadIdx.x;
  if (i < n) d[i] = (__bf16)s[i];
}

// ---------------- transpose-convert: src (K x N row-major f32) -> dst (N x K bf16) ----------------
__global__ void k_transpose_bf16(const float* __restrict__ src, __bf16* __restrict__ dst,
                                 int N, int K) {
  int i = blockIdx.x * blockDim.x + threadIdx.x;   // over N*K
  if (i >= N * K) return;
  int n = i / K, c = i - n * K;
  dst[i] = (__bf16)src[(size_t)c * N + n];
}

// ---------------- WMMA fragment loaders (contiguous k = 16*hv + s per lane, vectorized) ----------------
// A: 16x32 bf16 tile from row-major src; per-lane k-run is one contiguous 32B span.
__device__ __forceinline__ v16bf load_a(const __bf16* A, int lda, int m, int k0, int hv) {
  return *(const v16bf*)(A + (size_t)m * lda + k0 + hv * 16);
}
// B: 32x16 bf16 tile from N-major (transposed) src; per-lane k-run is one contiguous 32B span.
__device__ __forceinline__ v16bf load_bt(const __bf16* Bt, int K, int k0, int n, int hv) {
  return *(const v16bf*)(Bt + (size_t)n * K + k0 + hv * 16);
}

// ---------------- bf16 WMMA GEMM: C(MxN) = A(MxK, row-major) * Bt(NxK, N-major) ----------------
// One 16x32 output tile per wave (2 WMMA sharing the A fragment per k-step).
template <bool OUTF32>
__global__ void k_gemm(const __bf16* __restrict__ A, const __bf16* __restrict__ Bt,
                       void* __restrict__ Cv, int M, int N, int K) {
  int wave = (int)((blockIdx.x * blockDim.x + threadIdx.x) >> 5);
  int lane = threadIdx.x & 31;
  int tn2 = N >> 5;
  if (wave >= (M >> 4) * tn2) return;   // wave-uniform: EXEC all-1 for WMMA
  int tm = wave / tn2, tn = wave % tn2;
  int m0 = tm << 4, n0 = tn << 5;
  int hv = lane >> 4;
  int ml = m0 + (lane & 15);
  int nl = n0 + (lane & 15);
  v8f acc0 = {}, acc1 = {};
#pragma unroll 4
  for (int k0 = 0; k0 < K; k0 += 32) {
    if (k0 + 32 < K) {
      __builtin_prefetch(A + (size_t)ml * K + k0 + 32, 0, 0);
      __builtin_prefetch(Bt + (size_t)nl * K + k0 + 32, 0, 0);
    }
    v16bf a  = load_a(A, K, ml, k0, hv);
    v16bf b0 = load_bt(Bt, K, k0, nl, hv);
    v16bf b1 = load_bt(Bt, K, k0, nl + 16, hv);
    acc0 = __builtin_amdgcn_wmma_f32_16x16x32_bf16(false, a, false, b0, (short)0, acc0, false, false);
    acc1 = __builtin_amdgcn_wmma_f32_16x16x32_bf16(false, a, false, b1, (short)0, acc1, false, false);
  }
#pragma unroll
  for (int r = 0; r < 8; ++r) {
    int m = m0 + r + hv * 8;
    if constexpr (OUTF32) {
      float* C = (float*)Cv;
      C[(size_t)m * N + nl]      = acc0[r];
      C[(size_t)m * N + nl + 16] = acc1[r];
    } else {
      __bf16* C = (__bf16*)Cv;
      C[(size_t)m * N + nl]      = (__bf16)acc0[r];
      C[(size_t)m * N + nl + 16] = (__bf16)acc1[r];
    }
  }
}

// ---------------- indexer: S[t,j] = sum_h wI[t,h]*relu(qI_h . kI_j / sqrt(32)) ----------------
__global__ void k_indexer(const __bf16* __restrict__ qI, const __bf16* __restrict__ kI,
                          const float* __restrict__ wI, float* __restrict__ S) {
  int wave = (int)((blockIdx.x * blockDim.x + threadIdx.x) >> 5);
  int lane = threadIdx.x & 31;
  const int tiles = T_SEQ / 16;
  if (wave >= tiles * tiles) return;
  int tm = wave / tiles, tn = wave % tiles;
  int m0 = tm << 4, n0 = tn << 4;
  int hv = lane >> 4;
  int ml = m0 + (lane & 15);
  int nl = n0 + (lane & 15);
  // B fragment (kI is already N-major: T x 32) is identical for all 4 heads.
  v16bf b = *(const v16bf*)(kI + (size_t)nl * DI_N + hv * 16);
  float accS[8] = {};
#pragma unroll
  for (int h = 0; h < HI_N; ++h) {
    v16bf a = load_a(qI, HI_N * DI_N, ml, h * DI_N, hv);
    v8f t = {};
    t = __builtin_amdgcn_wmma_f32_16x16x32_bf16(false, a, false, b, (short)0, t, false, false);
#pragma unroll
    for (int r = 0; r < 8; ++r) {
      int m = m0 + r + hv * 8;
      float sv = fmaxf(t[r] * 0.17677669529663687f, 0.f);   // relu(x/sqrt(32))
      accS[r] += wI[m * HI_N + h] * sv;
    }
  }
#pragma unroll
  for (int r = 0; r < 8; ++r) {
    int m = m0 + r + hv * 8;
    S[(size_t)m * T_SEQ + nl] = accS[r];
  }
}

// ---------------- per-row stats + top-128 selection (async row load, bitonic sort) ----------------
__global__ void k_select(const float* __restrict__ S, int* __restrict__ topidx) {
  __shared__ float vals[T_SEQ];
  __shared__ int   idxs[T_SEQ];
  __shared__ float red[256];
  __shared__ float s_mean, s_kt;
  int t = blockIdx.x;
  int tid = threadIdx.x;
  const float* row = S + (size_t)t * T_SEQ;

  // async load: each thread pulls 4 contiguous f32 of the S row straight into LDS
  {
    unsigned int  ldsoff = (unsigned int)(uintptr_t)(&vals[tid * 4]);
    unsigned long long ga = (unsigned long long)(uintptr_t)(row + tid * 4);
    asm volatile("global_load_async_to_lds_b128 %0, %1, off" :: "v"(ldsoff), "v"(ga) : "memory");
    asm volatile("s_wait_asynccnt 0x0" ::: "memory");
  }
  __syncthreads();

  float ps = 0.f;
  for (int j = tid; j <= t; j += 256) ps += vals[j];
  red[tid] = ps; __syncthreads();
  for (int o = 128; o > 0; o >>= 1) { if (tid < o) red[tid] += red[tid + o]; __syncthreads(); }
  float cnt = (float)(t + 1);
  if (tid == 0) s_mean = red[0] / cnt;
  __syncthreads();
  float mean = s_mean;
  float pv = 0.f;
  for (int j = tid; j <= t; j += 256) { float d = vals[j] - mean; pv += d * d; }
  red[tid] = pv; __syncthreads();
  for (int o = 128; o > 0; o >>= 1) { if (tid < o) red[tid] += red[tid + o]; __syncthreads(); }
  if (tid == 0) {
    float var = red[0] / cnt;
    float kt = rintf((float)KBASE * var);
    kt = fminf(fmaxf(kt, (float)KMIN), (float)KMAX);
    s_kt = fminf(kt, cnt);
  }
  __syncthreads();
  // selection keys (causal mask + attention sinks)
  for (int j = tid; j < T_SEQ; j += 256) {
    float v = (j <= t) ? ((j < SINKN) ? 1e30f : vals[j]) : NEGV;
    vals[j] = v; idxs[j] = j;
  }
  __syncthreads();
  // bitonic sort, descending
  for (int ksz = 2; ksz <= T_SEQ; ksz <<= 1) {
    for (int jsz = ksz >> 1; jsz > 0; jsz >>= 1) {
      for (int i = tid; i < T_SEQ; i += 256) {
        if ((i & jsz) == 0) {
          int l = i | jsz;
          bool desc = ((i & ksz) == 0);
          float vi = vals[i], vl = vals[l];
          bool sw = desc ? (vi < vl) : (vi > vl);
          if (sw) {
            vals[i] = vl; vals[l] = vi;
            int ti = idxs[i]; idxs[i] = idxs[l]; idxs[l] = ti;
          }
        }
      }
      __syncthreads();
    }
  }
  if (tid < KMAX) {
    bool keep = ((float)tid < s_kt) && (vals[tid] > NEGV * 0.5f);
    topidx[t * KMAX + tid] = keep ? idxs[tid] : -1;
  }
}

// ---------------- YaRN RoPE on q,k + sigmoid gating of v (reads fused proj buffer) ----------------
__global__ void k_rope_gate(const float* __restrict__ proj,
                            __bf16* __restrict__ qr, __bf16* __restrict__ kr,
                            __bf16* __restrict__ vg) {
  int t = blockIdx.x;
  int tid = threadIdx.x;          // 256
  int h = tid >> 5;               // head 0..7
  int pr = tid & 31;              // pair 0..31
  int fi = (2 * pr) & 31;         // freqs index (emb[::2] pattern)
  float ar = (float)fi * (1.0f / 32.0f);
  float scaled_base = 10000.0f * __powf(32.0f, 64.0f / 62.0f);  // BASE * ext^(d/(d-2))
  float inv_freq   = __powf(scaled_base, -ar);
  float freq_extra = __powf(10000.0f, -ar);
  float wavelen = 6.283185307179586f / freq_extra;
  float ramp  = fminf(fmaxf((wavelen - 1.0f) / 31.0f, 0.0f), 1.0f);
  float scale = 1.0f + 31.0f * ramp;
  float f = ((float)t / scale) * inv_freq;
  float cs = __cosf(f), sn = __sinf(f);

  const float* rowp = proj + (size_t)t * NPROJ;
  const float* qf  = rowp;
  const float* kf  = rowp + C_DIM;
  const float* vf  = rowp + 2 * C_DIM;
  const float* gvf = rowp + 3 * C_DIM;

  size_t ob = (size_t)t * C_DIM + h * HD_N;
  int hb = h * HD_N;
  float x1 = qf[hb + 2 * pr], x2 = qf[hb + 2 * pr + 1];
  qr[ob + pr]      = (__bf16)(x1 * cs - x2 * sn);
  qr[ob + pr + 32] = (__bf16)(x1 * sn + x2 * cs);
  x1 = kf[hb + 2 * pr]; x2 = kf[hb + 2 * pr + 1];
  kr[ob + pr]      = (__bf16)(x1 * cs - x2 * sn);
  kr[ob + pr + 32] = (__bf16)(x1 * sn + x2 * cs);

  int d0 = 2 * pr;
  float g0 = 1.0f / (1.0f + __expf(-gvf[hb + d0]));
  float g1 = 1.0f / (1.0f + __expf(-gvf[hb + d0 + 1]));
  vg[ob + d0]     = (__bf16)(vf[hb + d0] * g0);
  vg[ob + d0 + 1] = (__bf16)(vf[hb + d0 + 1] * g1);
}

// ---------------- wI = sigmoid(x @ W_Iw + bias) ----------------
__global__ void k_wI(const float* __restrict__ x, const float* __restrict__ Wiw,
                     const float* __restrict__ gb, float* __restrict__ wI) {
  int tid = blockIdx.x * blockDim.x + threadIdx.x;
  if (tid >= T_SEQ * HI_N) return;
  int t = tid >> 2, hh = tid & 3;
  float sum = gb[hh];
  const float* xp = x + (size_t)t * C_DIM;
  for (int c = 0; c < C_DIM; ++c) sum += xp[c] * Wiw[c * HI_N + hh];
  wI[tid] = 1.0f / (1.0f + __expf(-sum));
}

// ---------------- sparse gathered attention (block per t, wave32 per head) ----------------
__global__ void k_attn(const __bf16* __restrict__ qr, const __bf16* __restrict__ kr,
                       const __bf16* __restrict__ vg, const float* __restrict__ proj,
                       const int* __restrict__ topidx, __bf16* __restrict__ ao) {
  __shared__ int   idx_sm[KMAX];
  __shared__ float p_sm[H_N][KMAX];
  __shared__ float q_sm[H_N][HD_N];
  int t = blockIdx.x;
  int tid = threadIdx.x;
  int h = tid >> 5, lane = tid & 31;
  if (tid < KMAX) idx_sm[tid] = topidx[t * KMAX + tid];
  q_sm[h][2 * lane]     = bf2f(qr[(size_t)t * C_DIM + h * HD_N + 2 * lane]);
  q_sm[h][2 * lane + 1] = bf2f(qr[(size_t)t * C_DIM + h * HD_N + 2 * lane + 1]);
  __syncthreads();

  float sc[4];
  float mx = NEGV;
#pragma unroll
  for (int ssl = 0; ssl < 4; ++ssl) {
    int slot = lane + 32 * ssl;
    int id = idx_sm[slot];
    float s = NEGV;
    if (id >= 0) {
      const v8bf* kp8 = (const v8bf*)(kr + (size_t)id * C_DIM + h * HD_N);
      float sum = 0.f;
#pragma unroll
      for (int c = 0; c < 8; ++c) {
        v8bf kv = kp8[c];
#pragma unroll
        for (int e = 0; e < 8; ++e) sum += q_sm[h][c * 8 + e] * bf2f(kv[e]);
      }
      s = sum * 0.125f;   // 1/sqrt(64)
    }
    sc[ssl] = s;
    mx = fmaxf(mx, s);
  }
#pragma unroll
  for (int o = 16; o > 0; o >>= 1) mx = fmaxf(mx, __shfl_xor(mx, o, 32));
  float se = 0.f, e4[4];
#pragma unroll
  for (int ssl = 0; ssl < 4; ++ssl) { e4[ssl] = __expf(sc[ssl] - mx); se += e4[ssl]; }
#pragma unroll
  for (int o = 16; o > 0; o >>= 1) se += __shfl_xor(se, o, 32);
  float inv = 1.0f / se;
#pragma unroll
  for (int ssl = 0; ssl < 4; ++ssl) p_sm[h][lane + 32 * ssl] = e4[ssl] * inv;
  __syncthreads();

  int d0 = 2 * lane, d1 = 2 * lane + 1;
  float a0 = 0.f, a1 = 0.f;
  for (int slot = 0; slot < KMAX; ++slot) {
    int id = idx_sm[slot];
    if (id >= 0) {
      float p = p_sm[h][slot];
      v2bf vv = *(const v2bf*)(vg + (size_t)id * C_DIM + h * HD_N + d0);
      a0 += p * bf2f(vv[0]);
      a1 += p * bf2f(vv[1]);
    }
  }
  size_t ob = (size_t)t * C_DIM + h * HD_N;
  const float* gof = proj + (size_t)t * NPROJ + 4 * C_DIM + h * HD_N;   // go slice
  float g0 = 1.0f / (1.0f + __expf(-gof[d0]));
  float g1 = 1.0f / (1.0f + __expf(-gof[d1]));
  ao[ob + d0] = (__bf16)(a0 * g0);
  ao[ob + d1] = (__bf16)(a1 * g1);
}

// ---------------- host orchestration ----------------
extern "C" void kernel_launch(void* const* d_in, const int* in_sizes, int n_in,
                              void* d_out, int out_size, void* d_ws, size_t ws_size,
                              hipStream_t stream) {
  (void)in_sizes; (void)n_in; (void)out_size; (void)ws_size;
  const float* x    = (const float*)d_in[0];
  const float* W_Iq = (const float*)d_in[1];
  const float* W_Ik = (const float*)d_in[2];
  const float* W_Iw = (const float*)d_in[3];
  const float* gb   = (const float*)d_in[4];
  const float* W_q  = (const float*)d_in[5];
  const float* W_k  = (const float*)d_in[6];
  const float* W_v  = (const float*)d_in[7];
  const float* W_gv = (const float*)d_in[8];
  const float* W_go = (const float*)d_in[9];
  const float* W_o  = (const float*)d_in[10];
  float* out = (float*)d_out;

  char* ws = (char*)d_ws;
  size_t off = 0;
  auto alloc = [&](size_t bytes) -> char* {
    char* p = ws + off;
    off += (bytes + 255) & ~(size_t)255;
    return p;
  };
  __bf16* xb    = (__bf16*)alloc((size_t)T_SEQ * C_DIM * 2);
  __bf16* WIqT  = (__bf16*)alloc((size_t)128 * C_DIM * 2);     // N-major
  __bf16* WIkT  = (__bf16*)alloc((size_t)32 * C_DIM * 2);      // N-major
  __bf16* WtAll = (__bf16*)alloc((size_t)NPROJ * C_DIM * 2);   // q|k|v|gv|go, N-major
  __bf16* WoT   = (__bf16*)alloc((size_t)C_DIM * C_DIM * 2);   // N-major
  __bf16* qIb   = (__bf16*)alloc((size_t)T_SEQ * 128 * 2);
  __bf16* kIb   = (__bf16*)alloc((size_t)T_SEQ * 32 * 2);
  float*  wI    = (float*)alloc((size_t)T_SEQ * HI_N * 4);
  float*  proj  = (float*)alloc((size_t)T_SEQ * NPROJ * 4);
  __bf16* qr    = (__bf16*)alloc((size_t)T_SEQ * C_DIM * 2);
  __bf16* kr    = (__bf16*)alloc((size_t)T_SEQ * C_DIM * 2);
  __bf16* vg    = (__bf16*)alloc((size_t)T_SEQ * C_DIM * 2);
  float*  S     = (float*)alloc((size_t)T_SEQ * T_SEQ * 4);
  int*    tidx  = (int*)alloc((size_t)T_SEQ * KMAX * 4);
  __bf16* ao    = (__bf16*)alloc((size_t)T_SEQ * C_DIM * 2);

  // activations: straight bf16 convert (row-major)
  k_f32_to_bf16<<<(T_SEQ * C_DIM + 255) / 256, 256, 0, stream>>>(x, xb, T_SEQ * C_DIM);
  // weights: transpose-convert to N-major bf16
  auto tr = [&](const float* s, __bf16* d, int N) {
    int n = N * C_DIM;
    k_transpose_bf16<<<(n + 255) / 256, 256, 0, stream>>>(s, d, N, C_DIM);
  };
  tr(W_Iq, WIqT, 128);
  tr(W_Ik, WIkT, 32);
  tr(W_q,  WtAll + 0 * (size_t)C_DIM * C_DIM, C_DIM);
  tr(W_k,  WtAll + 1 * (size_t)C_DIM * C_DIM, C_DIM);
  tr(W_v,  WtAll + 2 * (size_t)C_DIM * C_DIM, C_DIM);
  tr(W_gv, WtAll + 3 * (size_t)C_DIM * C_DIM, C_DIM);
  tr(W_go, WtAll + 4 * (size_t)C_DIM * C_DIM, C_DIM);
  tr(W_o,  WoT, C_DIM);

  auto gemm_f32 = [&](const __bf16* A, const __bf16* Bt, float* C, int M, int N, int K) {
    int waves = (M / 16) * (N / 32);
    k_gemm<true><<<(waves + 7) / 8, 256, 0, stream>>>(A, Bt, (void*)C, M, N, K);
  };
  auto gemm_bf16 = [&](const __bf16* A, const __bf16* Bt, __bf16* C, int M, int N, int K) {
    int waves = (M / 16) * (N / 32);
    k_gemm<false><<<(waves + 7) / 8, 256, 0, stream>>>(A, Bt, (void*)C, M, N, K);
  };

  // indexer projections + gate
  gemm_bf16(xb, WIqT, qIb, T_SEQ, 128, C_DIM);
  gemm_bf16(xb, WIkT, kIb, T_SEQ, 32, C_DIM);
  k_wI<<<(T_SEQ * HI_N + 255) / 256, 256, 0, stream>>>(x, W_Iw, gb, wI);

  // fused q|k|v|gv|go projection (single WMMA GEMM, N=2560)
  gemm_f32(xb, WtAll, proj, T_SEQ, NPROJ, C_DIM);

  // indexer score matrix S
  {
    int tiles = (T_SEQ / 16) * (T_SEQ / 16);
    k_indexer<<<(tiles + 7) / 8, 256, 0, stream>>>(qIb, kIb, wI, S);
  }

  // stats + top-k selection
  k_select<<<T_SEQ, 256, 0, stream>>>(S, tidx);

  // rope + v gating
  k_rope_gate<<<T_SEQ, 256, 0, stream>>>(proj, qr, kr, vg);

  // gathered attention + output gating
  k_attn<<<T_SEQ, 256, 0, stream>>>(qr, kr, vg, proj, tidx, ao);

  // final projection to d_out (f32)
  gemm_f32(ao, WoT, out, T_SEQ, C_DIM, C_DIM);
}
